// DSTA_70677981823326
// MI455X (gfx1250) — matrix operand order repaired and checked by reading, not compiled, not run on hardware
//
#include <hip/hip_runtime.h>
#include <hip/hip_bf16.h>

typedef _Float16 v16h __attribute__((ext_vector_type(16)));
typedef float    v8f  __attribute__((ext_vector_type(8)));

// ---------------------------------------------------------------------------
// Implicit-GEMM conv via V_WMMA_F32_16X16X32_F16 (fp32 accumulate).
//   out[b][o][oy][ox] = act( sum_{c,ky,kx} in[b][c][oy*s-p+ky][ox*s-p+kx] *
//                            w[o][c][ky][kx] + bias[o] )
// One wave = (MT*16 out-chan M) x (16 pixel N) tile, K = C*KH*KW in steps of
// 32.  Per K-step the gathered input fragment (B) is built ONCE and reused by
// MT WMMAs against MT weight fragments/accumulators -> MT x less input
// traffic and MT WMMAs amortize each address decode.
//
// K ordering is PERMUTED as k = (ky*KW + kx)*C + c  (valid: A and B use the
// same map).  C is a power of two for every conv in this net, so the decode
// is mask/shift only; for C>=32 the tap (ky,kx) is uniform per K-step.
//
// VGPR layouts per CDNA5 ISA 7.12.2:
//   A (16x32 f16): lane m=l&15; elem e (r=e>>1,t=e&1):
//                  K = e + 16*(l>>4) (this code's order, not the doc's table)
//   B (32x16 f16): lane n=l&15; elem e: K = e + 16*(l>>4) (this code's order, same as A)
//   D (16x16 f32): lane n=l&15; VGPR r -> M = r + 8*(l>>4)
// act: 0=none, 1=relu, 2=silu
// ---------------------------------------------------------------------------
template<int C, int H, int W, int O, int OH, int OW,
         int KH, int KW, int STRIDE, int PAD, int ACT, int MT>
__global__ __launch_bounds__(32)
void conv_wmma(const float* __restrict__ in, const float* __restrict__ wgt,
               const float* __restrict__ bias, float* __restrict__ out)
{
    static_assert((C & (C - 1)) == 0, "C must be a power of two");
    static_assert((O % (16 * MT)) == 0, "O must be a multiple of 16*MT");
    constexpr int KK    = KH * KW;
    constexpr int Ktot  = C * KK;
    constexpr int NSTEP = (Ktot + 31) / 32;
    constexpr int TILES_X = OW >> 4;

    const int lane = threadIdx.x;
    const int n    = lane & 15;
    const int hi   = lane >> 4;

    int gid = blockIdx.x;
    const int tx = gid % TILES_X; gid /= TILES_X;
    const int oy = gid % OH;      gid /= OH;
    const int b  = gid;
    const int ob = (blockIdx.y * MT) << 4;    // first out-channel of this wave

    const int ox  = (tx << 4) + n;            // pixel column (lane N dim)
    const int iy0 = oy * STRIDE - PAD;
    const int ix0 = ox * STRIDE - PAD;

    const float* __restrict__ inb = in + (size_t)b * C * H * W;
    // A rows: weight base for out-channel (ob + mt*16 + n)
    const float* __restrict__ wrow0 = wgt + (size_t)(ob + n) * Ktot;

    v8f acc[MT];
#pragma unroll
    for (int mt = 0; mt < MT; ++mt) acc[mt] = (v8f){};

#pragma unroll
    for (int s = 0; s < NSTEP; ++s) {
        const int k0 = s * 32;

        // ---- B fragment (input pixel ox), built once per step ----
        v16h bfr;
#pragma unroll
        for (int e = 0; e < 16; ++e) {
            const int kb = k0 + e + 16 * hi;
            float bv = 0.f;
            if (kb < Ktot) {
                const int cb = kb & (C - 1);
                const int tb = kb / C;             // uniform per step for C>=32
                const int ky = tb / KW, kx = tb % KW;   // constant divisor
                const int iy = iy0 + ky, ix = ix0 + kx;
                if ((unsigned)iy < (unsigned)H && (unsigned)ix < (unsigned)W)
                    bv = inb[((size_t)cb * H + iy) * W + ix];
            }
            bfr[e] = (_Float16)bv;
        }

        // ---- MT weight fragments x MT WMMAs reuse the same B fragment ----
#pragma unroll
        for (int mt = 0; mt < MT; ++mt) {
            const float* __restrict__ wrow = wrow0 + (size_t)(mt * 16) * Ktot;
            v16h afr;
#pragma unroll
            for (int e = 0; e < 16; ++e) {
                const int r = e >> 1, t = e & 1;
                const int koff = (r < 4) ? (2 * r + t) : (2 * (r - 4) + t + 16);
                const int ka = k0 + e + 16 * hi;
                float av = 0.f;
                if (ka < Ktot) {                   // folds unless Ktot%32 != 0
                    const int ca = ka & (C - 1);
                    const int ta = ka / C;         // power-of-2 -> shift
                    av = wrow[ca * KK + ta];       // w[o][ca][tap] (C-stride KK)
                }
                afr[e] = (_Float16)av;
            }
            acc[mt] = __builtin_amdgcn_wmma_f32_16x16x32_f16(
                /*neg_a=*/false, afr, /*neg_b=*/false, bfr,
                /*c_mod=*/(short)0, acc[mt], /*reuse_a=*/false, /*reuse_b=*/false);
        }
    }

#pragma unroll
    for (int mt = 0; mt < MT; ++mt) {
#pragma unroll
        for (int r = 0; r < 8; ++r) {
            const int ch = ob + mt * 16 + r + 8 * hi;  // D row M = out channel
            float v = acc[mt][r] + (bias ? bias[ch] : 0.f);
            if (ACT == 1)      v = fmaxf(v, 0.f);
            else if (ACT == 2) v = v * (1.f / (1.f + __expf(-v)));   // silu
            out[(((size_t)b * O + ch) * OH + oy) * OW + ox] = v;
        }
    }
}

// ---------------------------------------------------------------------------
// Per-pixel channel mean/max -> (B, 2, H, W)  [avg in ch0, max in ch1]
// ---------------------------------------------------------------------------
__global__ void chan_stats_kernel(const float* __restrict__ x2,
                                  float* __restrict__ amax,
                                  int B, int C, int HW)
{
    const int idx = blockIdx.x * blockDim.x + threadIdx.x;
    if (idx >= B * HW) return;
    const int b = idx / HW, p = idx - b * HW;
    const float* base = x2 + (size_t)b * C * HW + p;
    float s = 0.f, mx = -INFINITY;
    for (int c = 0; c < C; ++c) {
        const float v = base[(size_t)c * HW];
        s += v; mx = fmaxf(mx, v);
    }
    amax[((size_t)b * 2 + 0) * HW + p] = s / (float)C;
    amax[((size_t)b * 2 + 1) * HW + p] = mx;
}

// ---------------------------------------------------------------------------
// Global avg+max pool per (b,c) over H*W.  One 256-thread block per (b,c).
// ---------------------------------------------------------------------------
__global__ void global_pool_kernel(const float* __restrict__ x2,
                                   float* __restrict__ ap, float* __restrict__ mp,
                                   int HW)
{
    __shared__ float ssum[256], smax[256];
    const float* base = x2 + (size_t)blockIdx.x * HW;
    float s = 0.f, mx = -INFINITY;
    for (int i = threadIdx.x; i < HW; i += 256) {
        const float v = base[i];
        s += v; mx = fmaxf(mx, v);
    }
    ssum[threadIdx.x] = s; smax[threadIdx.x] = mx;
    __syncthreads();
    for (int off = 128; off > 0; off >>= 1) {
        if (threadIdx.x < off) {
            ssum[threadIdx.x] += ssum[threadIdx.x + off];
            smax[threadIdx.x] = fmaxf(smax[threadIdx.x], smax[threadIdx.x + off]);
        }
        __syncthreads();
    }
    if (threadIdx.x == 0) {
        ap[blockIdx.x] = ssum[0] / (float)HW;
        mp[blockIdx.x] = smax[0];
    }
}

// ---------------------------------------------------------------------------
// Channel-attention MLP:  scale = sigmoid( W2 @ (relu(W1@ap) + relu(W1@mp)) )
// grid = B, block = 32 (F=32, Fh=16)
// ---------------------------------------------------------------------------
__global__ void ca_mlp_kernel(const float* __restrict__ ap, const float* __restrict__ mp,
                              const float* __restrict__ w1, const float* __restrict__ w2,
                              float* __restrict__ scale, int F, int Fh)
{
    __shared__ float ha[32], hm[32];
    const int b = blockIdx.x, t = threadIdx.x;
    if (t < Fh) {
        float sa_ = 0.f, sm_ = 0.f;
        for (int c = 0; c < F; ++c) {
            const float w = w1[t * F + c];
            sa_ += w * ap[b * F + c];
            sm_ += w * mp[b * F + c];
        }
        ha[t] = fmaxf(sa_, 0.f);
        hm[t] = fmaxf(sm_, 0.f);
    }
    __syncthreads();
    if (t < F) {
        float s = 0.f;
        for (int h = 0; h < Fh; ++h) s += w2[t * Fh + h] * (ha[h] + hm[h]);
        scale[b * F + t] = 1.f / (1.f + __expf(-s));
    }
}

// ---------------------------------------------------------------------------
// cat = concat([x2_s, scale*x2], axis=channel)   (B, 2F, H, W)
// ---------------------------------------------------------------------------
__global__ void build_cat_kernel(const float* __restrict__ x2s, const float* __restrict__ x2,
                                 const float* __restrict__ scale, float* __restrict__ cat,
                                 int B, int F, int HW)
{
    const size_t idx = (size_t)blockIdx.x * blockDim.x + threadIdx.x;
    const size_t total = (size_t)B * 2 * F * HW;
    if (idx >= total) return;
    const int p = (int)(idx % HW);
    size_t t = idx / HW;
    const int ch = (int)(t % (2 * F));
    const int b  = (int)(t / (2 * F));
    float v;
    if (ch < F) v = x2s[((size_t)b * F + ch) * HW + p];
    else {
        const int c = ch - F;
        v = scale[b * F + c] * x2[((size_t)b * F + c) * HW + p];
    }
    cat[idx] = v;
}

// ---------------------------------------------------------------------------
// om1 += bilinear_upsample(om2, IHxIW -> OHxOW), half-pixel centers, clamped
// ---------------------------------------------------------------------------
__global__ void upsample_add_kernel(float* __restrict__ om1, const float* __restrict__ om2,
                                    int BC, int OH, int OW, int IH, int IW)
{
    const size_t idx = (size_t)blockIdx.x * blockDim.x + threadIdx.x;
    const size_t total = (size_t)BC * OH * OW;
    if (idx >= total) return;
    const int x = (int)(idx % OW);
    size_t t = idx / OW;
    const int y  = (int)(t % OH);
    const int bc = (int)(t / OH);

    const float sy = (y + 0.5f) * ((float)IH / (float)OH) - 0.5f;
    const float sx = (x + 0.5f) * ((float)IW / (float)OW) - 0.5f;
    const int y0 = (int)floorf(sy); const float wy = sy - (float)y0;
    const int x0 = (int)floorf(sx); const float wx = sx - (float)x0;
    const int y0c = min(max(y0, 0), IH - 1), y1c = min(max(y0 + 1, 0), IH - 1);
    const int x0c = min(max(x0, 0), IW - 1), x1c = min(max(x0 + 1, 0), IW - 1);

    const float* src = om2 + (size_t)bc * IH * IW;
    const float v = (1.f - wy) * ((1.f - wx) * src[y0c * IW + x0c] + wx * src[y0c * IW + x1c])
                  +         wy * ((1.f - wx) * src[y1c * IW + x0c] + wx * src[y1c * IW + x1c]);
    om1[idx] += v;
}

// ---------------------------------------------------------------------------
// Deformable conv v2 (per-channel offsets/masks), fused sigmoid on mask and
// relu+bias on output.  One thread per output pixel computes all 32 output
// channels; dcn_w (32*32*9 = 36 KB) staged in LDS.
//   off ch = c*18 + k*2 + {0,1};  mask ch = 18*F + c*9 + k   (inside om)
// ---------------------------------------------------------------------------
__global__ __launch_bounds__(256)
void dcn_v2_kernel(const float* __restrict__ x2f, const float* __restrict__ om,
                   const float* __restrict__ w, const float* __restrict__ bias,
                   float* __restrict__ out, int B, int F, int H, int W)
{
    __shared__ float wl[32 * 32 * 9];
    for (int i = threadIdx.x; i < 32 * 32 * 9; i += 256) wl[i] = w[i];
    __syncthreads();

    const size_t idx = (size_t)blockIdx.x * 256 + threadIdx.x;
    if (idx >= (size_t)B * H * W) return;
    const int x = (int)(idx % W);
    size_t t = idx / W;
    const int y = (int)(t % H);
    const int b = (int)(t / H);

    const size_t HW = (size_t)H * W;
    const size_t pix = (size_t)y * W + x;
    const float* xb  = x2f + (size_t)b * F * HW;
    const float* omb = om  + (size_t)b * 27 * F * HW;

    float acc[32];
#pragma unroll
    for (int o = 0; o < 32; ++o) acc[o] = 0.f;

    for (int c = 0; c < F; ++c) {
        const float* xc = xb + (size_t)c * HW;
        for (int k = 0; k < 9; ++k) {
            const float offy = omb[(size_t)(c * 18 + k * 2 + 0) * HW + pix];
            const float offx = omb[(size_t)(c * 18 + k * 2 + 1) * HW + pix];
            const float mraw = omb[(size_t)(18 * F + c * 9 + k) * HW + pix];
            const float m = 1.f / (1.f + __expf(-mraw));

            const float py = (float)y + (float)(k / 3 - 1) + offy;
            const float px = (float)x + (float)(k % 3 - 1) + offx;
            const float fy = floorf(py), fx = floorf(px);
            const float wy = py - fy,  wx = px - fx;
            const int y0 = (int)fy, x0 = (int)fx;

            float v00 = 0.f, v01 = 0.f, v10 = 0.f, v11 = 0.f;
            if ((unsigned)y0 < (unsigned)H) {
                if ((unsigned)x0 < (unsigned)W)       v00 = xc[(size_t)y0 * W + x0];
                if ((unsigned)(x0 + 1) < (unsigned)W) v01 = xc[(size_t)y0 * W + x0 + 1];
            }
            if ((unsigned)(y0 + 1) < (unsigned)H) {
                if ((unsigned)x0 < (unsigned)W)       v10 = xc[(size_t)(y0 + 1) * W + x0];
                if ((unsigned)(x0 + 1) < (unsigned)W) v11 = xc[(size_t)(y0 + 1) * W + x0 + 1];
            }
            const float val = (v00 * (1.f - wy) * (1.f - wx) + v01 * (1.f - wy) * wx +
                               v10 * wy * (1.f - wx) + v11 * wy * wx) * m;

            const float* wr = wl + c * 9 + k;
#pragma unroll
            for (int o = 0; o < 32; ++o) acc[o] += val * wr[o * 288];
        }
    }
#pragma unroll
    for (int o = 0; o < 32; ++o)
        out[((size_t)b * F + o) * HW + pix] = fmaxf(acc[o] + bias[o], 0.f);
}

// ---------------------------------------------------------------------------
static inline dim3 conv_grid(int B, int OH, int OW, int O, int MT) {
    return dim3((unsigned)((OW >> 4) * OH * B), (unsigned)(O / (16 * MT)), 1);
}

extern "C" void kernel_launch(void* const* d_in, const int* in_sizes, int n_in,
                              void* d_out, int out_size, void* d_ws, size_t ws_size,
                              hipStream_t stream)
{
    const float* x       = (const float*)d_in[0];
    const float* conv1_w = (const float*)d_in[1];
    const float* conv1_b = (const float*)d_in[2];
    const float* sa_w    = (const float*)d_in[3];
    const float* ca_w1   = (const float*)d_in[4];
    const float* ca_w2   = (const float*)d_in[5];
    const float* fuse_w  = (const float*)d_in[6];
    const float* fuse_b  = (const float*)d_in[7];
    const float* mask1_w = (const float*)d_in[8];
    const float* mask1_b = (const float*)d_in[9];
    const float* down_w  = (const float*)d_in[10];
    const float* down_b  = (const float*)d_in[11];
    const float* mask2_w = (const float*)d_in[12];
    const float* mask2_b = (const float*)d_in[13];
    const float* dcn_w   = (const float*)d_in[14];
    const float* dcn_b   = (const float*)d_in[15];
    const float* out_w   = (const float*)d_in[16];
    const float* out_b   = (const float*)d_in[17];
    float* outp = (float*)d_out;

    const int B = 4, N = 64, F = 32, H = 192, W = 192, Hd = 96, Wd = 96;
    const size_t HW = (size_t)H * W, HWd = (size_t)Hd * Wd;

    // ---- carve workspace (fp32) ----
    float* ws = (float*)d_ws;
    size_t o = 0;
    float* x2    = ws + o; o += (size_t)B * F * HW;       // conv1 out
    float* amax  = ws + o; o += (size_t)B * 2 * HW;       // [avg,max] over C
    float* x2s   = ws + o; o += (size_t)B * F * HW;       // silu(sa)
    float* ap    = ws + o; o += 128;
    float* mp    = ws + o; o += 128;
    float* scale = ws + o; o += 128;
    float* cat   = ws + o; o += (size_t)B * 2 * F * HW;   // [x2_s, x2_c]
    float* x2f   = ws + o; o += (size_t)B * F * HW;       // fused features
    float* om1   = ws + o; o += (size_t)B * 27 * F * HW;  // 510 MB, becomes om
    float* x3    = ws + o; o += (size_t)B * F * HWd;
    float* om2   = ws + o; o += (size_t)B * 27 * F * HWd;
    float* dcnout = x2;   // x2 is dead after build_cat -> reuse

    // 1) conv1 3x3 pad1, 64->32, relu   (MT=2: one wave does all 32 out-ch)
    conv_wmma<64, 192, 192, 32, 192, 192, 3, 3, 1, 1, 1, 2>
        <<<conv_grid(B, H, W, F, 2), 32, 0, stream>>>(x, conv1_w, conv1_b, x2);

    // 2) channel mean/max per pixel
    chan_stats_kernel<<<(unsigned)((B * HW + 255) / 256), 256, 0, stream>>>(
        x2, amax, B, F, (int)HW);

    // 3) spatial attention conv 7x7 pad3, 2->32, no bias, silu  -> x2_s
    conv_wmma<2, 192, 192, 32, 192, 192, 7, 7, 1, 3, 2, 2>
        <<<conv_grid(B, H, W, F, 2), 32, 0, stream>>>(amax, sa_w, nullptr, x2s);

    // 4) channel attention: global pools + MLP -> per-(b,c) sigmoid scale
    global_pool_kernel<<<B * F, 256, 0, stream>>>(x2, ap, mp, (int)HW);
    ca_mlp_kernel<<<B, 32, 0, stream>>>(ap, mp, ca_w1, ca_w2, scale, F, F / 2);

    // 5) cat = [x2_s, scale*x2]
    build_cat_kernel<<<(unsigned)(((size_t)B * 2 * F * HW + 255) / 256), 256, 0, stream>>>(
        x2s, x2, scale, cat, B, F, (int)HW);

    // 6) fuse 1x1, 64->32, relu -> x2f
    conv_wmma<64, 192, 192, 32, 192, 192, 1, 1, 1, 0, 1, 2>
        <<<conv_grid(B, H, W, F, 2), 32, 0, stream>>>(cat, fuse_w, fuse_b, x2f);

    // 7) mask1 3x3 pad1, 32->864 -> om1  (dominant GEMM: 147456 x 864 x 288)
    //    MT=6: each gathered input fragment feeds 6 WMMAs / 96 out-channels
    conv_wmma<32, 192, 192, 864, 192, 192, 3, 3, 1, 1, 0, 6>
        <<<conv_grid(B, H, W, 27 * F, 6), 32, 0, stream>>>(x2f, mask1_w, mask1_b, om1);

    // 8) down 3x3 stride2 pad1, 32->32, relu -> x3 (96x96)
    conv_wmma<32, 192, 192, 32, 96, 96, 3, 3, 2, 1, 1, 2>
        <<<conv_grid(B, Hd, Wd, F, 2), 32, 0, stream>>>(x2f, down_w, down_b, x3);

    // 9) mask2 3x3 pad1, 32->864 -> om2 (96x96), MT=6
    conv_wmma<32, 96, 96, 864, 96, 96, 3, 3, 1, 1, 0, 6>
        <<<conv_grid(B, Hd, Wd, 27 * F, 6), 32, 0, stream>>>(x3, mask2_w, mask2_b, om2);

    // 10) om = om1 + bilinear_up(om2)
    upsample_add_kernel<<<(unsigned)(((size_t)B * 27 * F * HW + 255) / 256), 256, 0, stream>>>(
        om1, om2, B * 27 * F, H, W, Hd, Wd);

    // 11) deformable conv v2 (+bias, relu) -> dcnout
    dcn_v2_kernel<<<(unsigned)((B * HW + 255) / 256), 256, 0, stream>>>(
        x2f, om1, dcn_w, dcn_b, dcnout, B, F, H, W);

    // 12) out conv 3x3 pad1, 32->64, relu -> d_out  (MT=4: all 64 out-ch)
    conv_wmma<32, 192, 192, 64, 192, 192, 3, 3, 1, 1, 1, 4>
        <<<conv_grid(B, H, W, N, 4), 32, 0, stream>>>(dcnout, out_w, out_b, outp);
}
